// Msingi1_87290915324349
// MI455X (gfx1250) — matrix operand order repaired and checked by reading, not compiled, not run on hardware
//
#include <hip/hip_runtime.h>
#include <hip/hip_bf16.h>
#include <math.h>

// ---------------------------------------------------------------------------
// Model constants (from reference)
// ---------------------------------------------------------------------------
constexpr int BATCH  = 2;
constexpr int SEQ    = 2048;
constexpr int TOK    = BATCH * SEQ;      // 4096
constexpr int EMBD   = 768;
constexpr int NH     = 12;
constexpr int HD     = 64;
constexpr int FFD    = 3072;
constexpr int NLAYER = 12;
constexpr int VOC    = 32000;

typedef __attribute__((ext_vector_type(16))) __bf16 v16bf;
typedef __attribute__((ext_vector_type(8)))  __bf16 v8bf;
typedef __attribute__((ext_vector_type(8)))  float  v8f;
typedef long long ll;

// f32 -> bf16 round-to-nearest-even, bit-level
static __device__ __forceinline__ __bf16 f2bf(float f) {
    union { float f; unsigned u; } x; x.f = f;
    unsigned r = x.u + 0x7fffu + ((x.u >> 16) & 1u);
    unsigned short hbits = (unsigned short)(r >> 16);
    __bf16 out;
    __builtin_memcpy(&out, &hbits, 2);
    return out;
}

static __device__ __forceinline__ v16bf cat8(v8bf a, v8bf b) {
    return __builtin_shufflevector(a, b, 0, 1, 2, 3, 4, 5, 6, 7,
                                         8, 9, 10, 11, 12, 13, 14, 15);
}

static __device__ __forceinline__ float gelu_f(float x) {
    float x3 = x * x * x;
    return 0.5f * x * (1.0f + tanhf(0.79788456f * (x + 0.044715f * x3)));
}

// ---------------------------------------------------------------------------
// Generic WMMA GEMM: C[M,N] = epilogue( A[M,K] * B )  (A f32 row-major,
// B f32 with element (k,n) at B[k*sBk + n*sBn]; both converted bf16 in LDS).
// Tile 128x64, BK=32, 256 threads = 8 waves; each wave owns a 2x2 grid of
// 16x16 f32 accumulators -> 4 v_wmma_f32_16x16x32_bf16 per K-step.
// Epilogue: + bias[col] (if bias), GELU (template), + res[row,col] (if res).
// M % 128 == 0, N % 64 == 0, K % 32 == 0 (true for all GEMMs here).
// ---------------------------------------------------------------------------
template <bool GELU>
__global__ __launch_bounds__(256)
void gemm_wmma_bf16(const float* __restrict__ A, const float* __restrict__ B,
                    float* __restrict__ C, const float* __restrict__ bias,
                    const float* __restrict__ res,
                    int M, int N, int K, ll sBk, ll sBn)
{
    constexpr int BM = 128, BN = 64, BK = 32;
    __shared__ __bf16 As[BM][BK];   // 8 KB
    __shared__ __bf16 Bt[BN][BK];   // 4 KB, transposed: Bt[n][k]

    const int tile_m = blockIdx.y * BM;
    const int tile_n = blockIdx.x * BN;
    const int tid  = threadIdx.x;
    const int wave = tid >> 5;
    const int lane = tid & 31;
    const int hb   = lane >> 4;     // lane half (0/1)
    const int l16  = lane & 15;

    const int mr0 = (wave & 3) * 2;  // first A subtile row (0,2,4,6)
    const int nc0 = (wave >> 2) * 2; // first B subtile col (0 or 2)

    v8f acc00 = {}, acc01 = {}, acc10 = {}, acc11 = {};

    // per-thread load mappings
    const int ar = tid >> 1;               // A row within tile (16 elems/thread)
    const int ac = (tid & 1) * 16;         // A col start
    const int bk = tid >> 3;               // B k within tile (8 elems/thread)
    const int bn = (tid & 7) * 8;          // B n start

    for (int k0 = 0; k0 < K; k0 += BK) {
        // ---- stage A (f32 -> bf16) ----
        {
            const float* src = A + (ll)(tile_m + ar) * K + (k0 + ac);
            if (k0 + BK < K) __builtin_prefetch(src + BK, 0, 1);
#pragma unroll
            for (int i = 0; i < 16; ++i) As[ar][ac + i] = f2bf(src[i]);
        }
        // ---- stage B transposed (f32 -> bf16) ----
        {
            const float* src = B + (ll)(k0 + bk) * sBk + (ll)(tile_n + bn) * sBn;
            if (k0 + BK < K) __builtin_prefetch(src + (ll)BK * sBk, 0, 1);
#pragma unroll
            for (int i = 0; i < 8; ++i) Bt[bn + i][bk] = f2bf(src[(ll)i * sBn]);
        }
        __syncthreads();

        // ---- fragments (explicit aligned vector LDS reads) ----
        const int rowA0 = mr0 * 16 + l16;
        const int rowA1 = rowA0 + 16;
        v16bf af0 = cat8(*(const v8bf*)&As[rowA0][hb * 8],
                         *(const v8bf*)&As[rowA0][16 + hb * 8]);
        v16bf af1 = cat8(*(const v8bf*)&As[rowA1][hb * 8],
                         *(const v8bf*)&As[rowA1][16 + hb * 8]);
        v16bf bf0 = cat8(*(const v8bf*)&Bt[nc0 * 16 + l16][hb * 8],
                         *(const v8bf*)&Bt[nc0 * 16 + l16][16 + hb * 8]);
        v16bf bf1 = cat8(*(const v8bf*)&Bt[(nc0 + 1) * 16 + l16][hb * 8],
                         *(const v8bf*)&Bt[(nc0 + 1) * 16 + l16][16 + hb * 8]);

        acc00 = __builtin_amdgcn_wmma_f32_16x16x32_bf16(false, af0, false, bf0,
                                                        (short)0, acc00, false, false);
        acc01 = __builtin_amdgcn_wmma_f32_16x16x32_bf16(false, af0, false, bf1,
                                                        (short)0, acc01, false, false);
        acc10 = __builtin_amdgcn_wmma_f32_16x16x32_bf16(false, af1, false, bf0,
                                                        (short)0, acc10, false, false);
        acc11 = __builtin_amdgcn_wmma_f32_16x16x32_bf16(false, af1, false, bf1,
                                                        (short)0, acc11, false, false);
        __syncthreads();
    }

    // ---- epilogue: C/D layout row = r + 8*hb, col = l16 within subtile ----
#pragma unroll
    for (int mi = 0; mi < 2; ++mi) {
        const int row0 = tile_m + (mr0 + mi) * 16 + hb * 8;
#pragma unroll
        for (int t = 0; t < 2; ++t) {
            v8f acc = (mi == 0) ? (t ? acc01 : acc00) : (t ? acc11 : acc10);
            const int col = tile_n + (nc0 + t) * 16 + l16;
            const float bval = bias ? bias[col] : 0.0f;
#pragma unroll
            for (int r = 0; r < 8; ++r) {
                float v = acc[r] + bval;
                if (GELU) v = gelu_f(v);
                if (res) v += res[(ll)(row0 + r) * N + col];
                C[(ll)(row0 + r) * N + col] = v;
            }
        }
    }
}

// ---------------------------------------------------------------------------
// Embedding gather: h[t, e] = emb[ids[t], e]
// ---------------------------------------------------------------------------
__global__ __launch_bounds__(256)
void embed_k(const int* __restrict__ ids, const float* __restrict__ emb,
             float* __restrict__ h)
{
    ll idx = (ll)blockIdx.x * blockDim.x + threadIdx.x;
    if (idx >= (ll)TOK * EMBD) return;
    int t = (int)(idx / EMBD);
    int e = (int)(idx % EMBD);
    h[idx] = emb[(ll)ids[t] * EMBD + e];
}

// ---------------------------------------------------------------------------
// LayerNorm over last dim (768); one block per token
// ---------------------------------------------------------------------------
__global__ __launch_bounds__(256)
void layernorm_k(const float* __restrict__ x, const float* __restrict__ sc,
                 const float* __restrict__ bi, float* __restrict__ y)
{
    __shared__ float red[256];
    const int t   = blockIdx.x;
    const int tid = threadIdx.x;
    const float* row = x + (ll)t * EMBD;

    float s = 0.0f;
    for (int i = tid; i < EMBD; i += 256) s += row[i];
    red[tid] = s; __syncthreads();
    for (int o = 128; o > 0; o >>= 1) { if (tid < o) red[tid] += red[tid + o]; __syncthreads(); }
    const float mu = red[0] * (1.0f / EMBD);
    __syncthreads();

    float vs = 0.0f;
    for (int i = tid; i < EMBD; i += 256) { float d = row[i] - mu; vs += d * d; }
    red[tid] = vs; __syncthreads();
    for (int o = 128; o > 0; o >>= 1) { if (tid < o) red[tid] += red[tid + o]; __syncthreads(); }
    const float rstd = rsqrtf(red[0] * (1.0f / EMBD) + 1e-5f);

    for (int i = tid; i < EMBD; i += 256)
        y[(ll)t * EMBD + i] = (row[i] - mu) * rstd * sc[i] + bi[i];
}

// ---------------------------------------------------------------------------
// RoPE on q and k in-place (f32, [TOK, EMBD] with col = h*64 + d)
// ---------------------------------------------------------------------------
__global__ __launch_bounds__(256)
void rope_k(float* __restrict__ q, float* __restrict__ k)
{
    ll idx = (ll)blockIdx.x * blockDim.x + threadIdx.x;
    if (idx >= (ll)TOK * NH * (HD / 2)) return;
    int i   = (int)(idx % (HD / 2));
    int tmp = (int)(idx / (HD / 2));
    int hh  = tmp % NH;
    int t   = tmp / NH;
    int pos = t % SEQ;
    // freq = theta^(-2i/HD); ln(10000) = 9.210340372
    float freq = __expf(-(float)(2 * i) * (1.0f / HD) * 9.210340372f);
    float ang  = (float)pos * freq;
    float c = __cosf(ang), s = __sinf(ang);
    ll base = (ll)t * EMBD + hh * HD + 2 * i;
    float qr = q[base], qi = q[base + 1];
    q[base]     = qr * c - qi * s;
    q[base + 1] = qr * s + qi * c;
    float kr = k[base], ki = k[base + 1];
    k[base]     = kr * c - ki * s;
    k[base + 1] = kr * s + ki * c;
}

// ---------------------------------------------------------------------------
// f32 -> bf16 flat convert
// ---------------------------------------------------------------------------
__global__ __launch_bounds__(256)
void cvt_bf16_k(const float* __restrict__ x, __bf16* __restrict__ y, ll n)
{
    ll idx = (ll)blockIdx.x * blockDim.x + threadIdx.x;
    if (idx < n) y[idx] = f2bf(x[idx]);
}

// ---------------------------------------------------------------------------
// V transpose+convert: vT[(b*NH+h)*HD + d][s] = v[(b*SEQ+s)*EMBD + h*HD + d]
// ---------------------------------------------------------------------------
__global__ __launch_bounds__(256)
void vtrans_k(const float* __restrict__ v, __bf16* __restrict__ vT)
{
    ll idx = (ll)blockIdx.x * blockDim.x + threadIdx.x;
    if (idx >= (ll)BATCH * NH * HD * SEQ) return;
    int s   = (int)(idx % SEQ);
    int tmp = (int)(idx / SEQ);
    int d   = tmp % HD;
    int bh  = tmp / HD;
    int hh  = bh % NH;
    int b   = bh / NH;
    vT[idx] = f2bf(v[(ll)(b * SEQ + s) * EMBD + hh * HD + d]);
}

// ---------------------------------------------------------------------------
// Flash attention (no mask, matches reference full softmax).
// One wave per (b, h, 16-query block); 32-key blocks.
// scores = Q(16x64) x K^T via wmma bf16; online softmax in f32 (half-wave
// shuffles); P re-shaped C-layout -> A-layout through 1KB LDS; O += P x V.
// ---------------------------------------------------------------------------
__global__ __launch_bounds__(32)
void attn_k(const __bf16* __restrict__ qb, const __bf16* __restrict__ kb,
            const __bf16* __restrict__ vT, float* __restrict__ out, float scale)
{
    __shared__ __bf16 P[16][32];

    const int gid  = blockIdx.x;
    const int qblk = gid % (SEQ / 16);
    const int bh   = gid / (SEQ / 16);
    const int hh   = bh % NH;
    const int b    = bh / NH;
    const int lane = threadIdx.x;
    const int hb   = lane >> 4;
    const int l16  = lane & 15;

    // Q fragments (A layout, 16x32 each): chunk0 d=0..31, chunk1 d=32..63
    const v8bf* qv = (const v8bf*)(qb + (ll)(b * SEQ + qblk * 16 + l16) * EMBD
                                      + hh * HD);
    v16bf qa0 = cat8(qv[hb],     qv[2 + hb]);
    v16bf qa1 = cat8(qv[4 + hb], qv[6 + hb]);

    float mrow[8], lsum[8];
#pragma unroll
    for (int r = 0; r < 8; ++r) { mrow[r] = -INFINITY; lsum[r] = 0.0f; }
    v8f o0 = {}, o1 = {}, o2 = {}, o3 = {};

    for (int jb = 0; jb < SEQ / 32; ++jb) {
        // ---- scores: two 16x16 fragments (keys 0..15 and 16..31 of block) ----
        v8f s0 = {}, s1 = {};
#pragma unroll
        for (int t = 0; t < 2; ++t) {
            const __bf16* krow =
                kb + (ll)(b * SEQ + jb * 32 + t * 16 + l16) * EMBD + hh * HD;
            v16bf kf0 = cat8(*(const v8bf*)&krow[hb * 8], *(const v8bf*)&krow[16 + hb * 8]);        // features 0..31
            v16bf kf1 = cat8(*(const v8bf*)&krow[32 + hb * 8], *(const v8bf*)&krow[48 + hb * 8]);   // features 32..63
            if (t == 0) {
                s0 = __builtin_amdgcn_wmma_f32_16x16x32_bf16(false, qa0, false, kf0,
                                                             (short)0, s0, false, false);
                s0 = __builtin_amdgcn_wmma_f32_16x16x32_bf16(false, qa1, false, kf1,
                                                             (short)0, s0, false, false);
            } else {
                s1 = __builtin_amdgcn_wmma_f32_16x16x32_bf16(false, qa0, false, kf0,
                                                             (short)0, s1, false, false);
                s1 = __builtin_amdgcn_wmma_f32_16x16x32_bf16(false, qa1, false, kf1,
                                                             (short)0, s1, false, false);
            }
        }

        // ---- online softmax per row (row = r + 8*hb, shared by 16-lane half) ----
#pragma unroll
        for (int r = 0; r < 8; ++r) {
            float v0 = s0[r] * scale;
            float v1 = s1[r] * scale;
            float mx = fmaxf(v0, v1);
#pragma unroll
            for (int off = 1; off < 16; off <<= 1)
                mx = fmaxf(mx, __shfl_xor(mx, off, 32));
            float mnew  = fmaxf(mrow[r], mx);
            float alpha = __expf(mrow[r] - mnew);
            mrow[r] = mnew;
            float p0 = __expf(v0 - mnew);
            float p1 = __expf(v1 - mnew);
            float ps = p0 + p1;
#pragma unroll
            for (int off = 1; off < 16; off <<= 1)
                ps += __shfl_xor(ps, off, 32);
            lsum[r] = lsum[r] * alpha + ps;
            o0[r] *= alpha; o1[r] *= alpha; o2[r] *= alpha; o3[r] *= alpha;
            P[r + 8 * hb][l16]      = f2bf(p0);
            P[r + 8 * hb][16 + l16] = f2bf(p1);
        }
        __syncthreads();

        // ---- P as A-fragment (16x32) ----
        v16bf pa = cat8(*(const v8bf*)&P[l16][hb * 8],
                        *(const v8bf*)&P[l16][16 + hb * 8]);
        __syncthreads();

        // ---- O += P x V : V block 32x64 via vT (feature-major rows) ----
#pragma unroll
        for (int nc = 0; nc < 4; ++nc) {
            v16bf vf = cat8(*(const v8bf*)(vT + (ll)(bh * HD + nc * 16 + l16) * SEQ
                                          + jb * 32 + hb * 8),
                            *(const v8bf*)(vT + (ll)(bh * HD + nc * 16 + l16) * SEQ
                                          + jb * 32 + 16 + hb * 8));
            if (nc == 0)      o0 = __builtin_amdgcn_wmma_f32_16x16x32_bf16(false, pa, false, vf, (short)0, o0, false, false);
            else if (nc == 1) o1 = __builtin_amdgcn_wmma_f32_16x16x32_bf16(false, pa, false, vf, (short)0, o1, false, false);
            else if (nc == 2) o2 = __builtin_amdgcn_wmma_f32_16x16x32_bf16(false, pa, false, vf, (short)0, o2, false, false);
            else              o3 = __builtin_amdgcn_wmma_f32_16x16x32_bf16(false, pa, false, vf, (short)0, o3, false, false);
        }
    }

    // ---- write out: out[(b*SEQ + q), h*64 + d], q = qblk*16 + r + 8*hb ----
#pragma unroll
    for (int r = 0; r < 8; ++r) {
        float inv = 1.0f / lsum[r];
        ll base = (ll)(b * SEQ + qblk * 16 + 8 * hb + r) * EMBD + hh * HD;
        out[base +  0 + l16] = o0[r] * inv;
        out[base + 16 + l16] = o1[r] * inv;
        out[base + 32 + l16] = o2[r] * inv;
        out[base + 48 + l16] = o3[r] * inv;
    }
}

// ---------------------------------------------------------------------------
// Host-side orchestration
// ---------------------------------------------------------------------------
static inline size_t align_up(size_t x, size_t a) { return (x + a - 1) & ~(a - 1); }

extern "C" void kernel_launch(void* const* d_in, const int* in_sizes, int n_in,
                              void* d_out, int out_size, void* d_ws, size_t ws_size,
                              hipStream_t stream)
{
    (void)in_sizes; (void)n_in; (void)out_size; (void)ws_size;

    const int*   ids   = (const int*)  d_in[0];
    const float* emb   = (const float*)d_in[1];
    const float* qw    = (const float*)d_in[2];
    const float* kw    = (const float*)d_in[3];
    const float* vw    = (const float*)d_in[4];
    const float* ow    = (const float*)d_in[5];
    const float* w1    = (const float*)d_in[6];
    const float* b1    = (const float*)d_in[7];
    const float* w2    = (const float*)d_in[8];
    const float* b2    = (const float*)d_in[9];
    const float* ln1s  = (const float*)d_in[10];
    const float* ln1b  = (const float*)d_in[11];
    const float* ln2s  = (const float*)d_in[12];
    const float* ln2b  = (const float*)d_in[13];
    const float* lnf_s = (const float*)d_in[14];
    const float* lnf_b = (const float*)d_in[15];
    float* logits = (float*)d_out;

    // -------- workspace carve-up --------
    char* ws = (char*)d_ws;
    size_t off = 0;
    const size_t szTE  = (size_t)TOK * EMBD * sizeof(float);     // 12.6 MB
    const size_t szTF  = (size_t)TOK * FFD  * sizeof(float);     // 50.3 MB
    const size_t szTEb = (size_t)TOK * EMBD * sizeof(__bf16);    // 6.3 MB

    float*  h      = (float*)(ws + off); off = align_up(off + szTE, 256);
    float*  x      = (float*)(ws + off); off = align_up(off + szTE, 256);
    float*  q      = (float*)(ws + off); off = align_up(off + szTE, 256);
    float*  k      = (float*)(ws + off); off = align_up(off + szTE, 256);
    float*  v      = (float*)(ws + off); off = align_up(off + szTE, 256);
    float*  ao     = (float*)(ws + off); off = align_up(off + szTE, 256);
    float*  a1     = (float*)(ws + off); off = align_up(off + szTF, 256);
    __bf16* qb     = (__bf16*)(ws + off); off = align_up(off + szTEb, 256);
    __bf16* kbuf   = (__bf16*)(ws + off); off = align_up(off + szTEb, 256);
    __bf16* vT     = (__bf16*)(ws + off); off = align_up(off + szTEb, 256);

    const float scale = 0.125f;  // HEAD_DIM^-0.5
    const dim3 blk256(256);
    const dim3 gE((TOK * EMBD + 255) / 256);
    const dim3 gRope(((ll)TOK * NH * (HD / 2) + 255) / 256);
    const dim3 gCvt(((ll)TOK * EMBD + 255) / 256);
    const dim3 gLN(TOK);
    const dim3 gAttn(BATCH * NH * (SEQ / 16));
    const dim3 gGemmE(EMBD / 64, TOK / 128);   // N=768
    const dim3 gGemmF(FFD  / 64, TOK / 128);   // N=3072
    const dim3 gGemmV(VOC  / 64, TOK / 128);   // N=32000

    // -------- embedding --------
    embed_k<<<gE, blk256, 0, stream>>>(ids, emb, h);

    for (int l = 0; l < NLAYER; ++l) {
        const float* Qw  = qw + (ll)l * EMBD * EMBD;
        const float* Kw  = kw + (ll)l * EMBD * EMBD;
        const float* Vw  = vw + (ll)l * EMBD * EMBD;
        const float* Ow  = ow + (ll)l * EMBD * EMBD;
        const float* W1  = w1 + (ll)l * EMBD * FFD;
        const float* B1  = b1 + (ll)l * FFD;
        const float* W2  = w2 + (ll)l * FFD * EMBD;
        const float* B2  = b2 + (ll)l * EMBD;

        // x = LN1(h)
        layernorm_k<<<gLN, blk256, 0, stream>>>(h, ln1s + (ll)l * EMBD,
                                                ln1b + (ll)l * EMBD, x);
        // q,k,v = x @ {Qw,Kw,Vw}
        gemm_wmma_bf16<false><<<gGemmE, blk256, 0, stream>>>(
            x, Qw, q, nullptr, nullptr, TOK, EMBD, EMBD, (ll)EMBD, 1LL);
        gemm_wmma_bf16<false><<<gGemmE, blk256, 0, stream>>>(
            x, Kw, k, nullptr, nullptr, TOK, EMBD, EMBD, (ll)EMBD, 1LL);
        gemm_wmma_bf16<false><<<gGemmE, blk256, 0, stream>>>(
            x, Vw, v, nullptr, nullptr, TOK, EMBD, EMBD, (ll)EMBD, 1LL);
        // RoPE on q,k
        rope_k<<<gRope, blk256, 0, stream>>>(q, k);
        // bf16 staging for attention
        cvt_bf16_k<<<gCvt, blk256, 0, stream>>>(q, qb,   (ll)TOK * EMBD);
        cvt_bf16_k<<<gCvt, blk256, 0, stream>>>(k, kbuf, (ll)TOK * EMBD);
        vtrans_k  <<<gCvt, blk256, 0, stream>>>(v, vT);
        // attention
        attn_k<<<gAttn, dim3(32), 0, stream>>>(qb, kbuf, vT, ao, scale);
        // h = h + ao @ Ow
        gemm_wmma_bf16<false><<<gGemmE, blk256, 0, stream>>>(
            ao, Ow, h, nullptr, h, TOK, EMBD, EMBD, (ll)EMBD, 1LL);
        // x = LN2(h)
        layernorm_k<<<gLN, blk256, 0, stream>>>(h, ln2s + (ll)l * EMBD,
                                                ln2b + (ll)l * EMBD, x);
        // a1 = gelu(x @ W1 + B1)
        gemm_wmma_bf16<true><<<gGemmF, blk256, 0, stream>>>(
            x, W1, a1, B1, nullptr, TOK, FFD, EMBD, (ll)FFD, 1LL);
        // h = h + a1 @ W2 + B2
        gemm_wmma_bf16<false><<<gGemmE, blk256, 0, stream>>>(
            a1, W2, h, B2, h, TOK, EMBD, FFD, (ll)EMBD, 1LL);
    }

    // final LN + tied-embedding logits: logits = LNf(h) @ emb.T
    layernorm_k<<<gLN, blk256, 0, stream>>>(h, lnf_s, lnf_b, x);
    gemm_wmma_bf16<false><<<gGemmV, blk256, 0, stream>>>(
        x, emb, logits, nullptr, nullptr, TOK, VOC, EMBD, 1LL, (ll)EMBD);
}
